// Decoder_64055142252682
// MI455X (gfx1250) — matrix operand order repaired and checked by reading, not compiled, not run on hardware
//
#include <hip/hip_runtime.h>

// ---------------- types ----------------
typedef __attribute__((ext_vector_type(16))) __bf16 v16bf;
typedef __attribute__((ext_vector_type(8)))  float  v8f;

// ---------------- problem dims ----------------
constexpr int kB    = 64;
constexpr int kTIN  = 256;
constexpr int kTOUT = 800;
constexpr int kENC  = 512;
constexpr int kATT  = 128;
constexpr int kNF   = 32;
constexpr int kKS   = 31;
constexpr int kPAD  = 15;
constexpr int kARNN = 1024;
constexpr int kPRE  = 256;
constexpr int kNMEL = 80;

// GEMM shapes (K always multiple of 32)
constexpr int KATT    = 1792;              // [x(256)|ctx(512)|ah(1024)]
constexpr int KC_ATT  = KATT / 32;         // 56
constexpr int KDEC    = 2560;              // [ah(1024)|ctx(512)|dh(1024)]
constexpr int KC_DEC  = KDEC / 32;         // 80
constexpr int NGATE   = 4096;
constexpr int NT_GATE = NGATE / 16;        // 256
constexpr int KC_PROJ = 1536 / 32;         // 48
constexpr int NPROJ   = 96;                // 80 mel + 1 gate + pad
constexpr int NT_PROJ = NPROJ / 16;        // 6
constexpr int KC_Q    = 1024 / 32;         // 32
constexpr int NT_Q    = kATT / 16;         // 8

// prenet GEMMs
constexpr int MPRE   = kTOUT * kB;         // 51200
constexpr int MT_PRE = MPRE / 16;          // 3200
constexpr int KC_P1  = 96 / 32;            // 3 (K=80 padded to 96)
constexpr int NT_P1  = kPRE / 16;          // 16
constexpr int KC_P2  = kPRE / 32;          // 8
constexpr int NT_P2  = kPRE / 16;          // 16

// processed_memory GEMM
constexpr int MPM   = kB * kTIN;           // 16384
constexpr int MT_PM = MPM / 16;            // 1024
constexpr int KC_PM = kENC / 32;           // 16
constexpr int NT_PM = kATT / 16;           // 8

// persistent grid
constexpr int NBLK  = 32;
constexpr int NTHR  = 256;
constexpr int NWAVE = NBLK * NTHR / 32;    // 256 waves
constexpr int XSTEP = 8 * 4 * 32 * 16;     // elements per step in Xpack (8 kchunks, MT=4)

// ---------------- helpers ----------------
__device__ __forceinline__ unsigned short f2bf(float f) {
  unsigned u = __float_as_uint(f);
  u += 0x7FFFu + ((u >> 16) & 1u);
  return (unsigned short)(u >> 16);
}
__device__ __forceinline__ float sigf(float x) { return 1.0f / (1.0f + __expf(-x)); }

// A-fragment (16x32 bf16, M x K) element offset, kchunk-major over MT tiles.
// Layout per CDNA5 ISA: lanes 0-15 hold K 0-7 (v0-3) & 16-23 (v4-7);
// lanes 16-31 hold K 8-15 & 24-31.
__device__ __forceinline__ size_t apack_off(int m, int k, int MT) {
  int mt = m >> 4, mi = m & 15;
  int kc = k >> 5, kin = k & 31;
  int hi = kin >> 4;            // 0 -> vgpr 0..3, 1 -> vgpr 4..7
  int t2 = kin & 15;
  int h  = t2 >> 3;             // lane half
  int w  = t2 & 7;
  int lane = h * 16 + mi;
  int elem = (hi * 4 + (w >> 1)) * 2 + (w & 1);
  return (((size_t)kc * MT + mt) * 32 + lane) * 16 + elem;
}

// B-fragment (32x16 bf16, K x N) element offset, ntile-major.
// lane = ((k/8)%2)*16 + n%16 ; elem = (k/16 within chunk)*8 + k%8.
__device__ __forceinline__ size_t bpack_off(int k, int n, int KC) {
  int nt = n >> 4;
  int kc = k >> 5, kin = k & 31;
  int lane = ((kin >> 3) & 1) * 16 + (n & 15);
  return (((size_t)nt * KC + kc) * 32 + lane) * 16 + (kin >> 4) * 8 + (kin & 7);
}

__device__ __forceinline__ v8f wmma_bf16(v8f c, v16bf a, v16bf b) {
  return __builtin_amdgcn_wmma_f32_16x16x32_bf16(false, a, false, b, (short)0, c,
                                                 false, false);
}

__device__ __forceinline__ v16bf ldfrag(const unsigned short* p) {
  return *(const v16bf*)p;
}

// ---------------- setup kernels ----------------
__global__ void k_zero(float* p, size_t nwords) {
  size_t i = (size_t)blockIdx.x * blockDim.x + threadIdx.x;
  size_t stride = (size_t)gridDim.x * blockDim.x;
  for (; i < nwords; i += stride) p[i] = 0.0f;
}

__global__ void k_pack_weights(const float* wih_a, const float* whh_a,
                               const float* wih_d, const float* whh_d,
                               const float* wq, const float* wm,
                               const float* pw1, const float* pw2,
                               const float* wproj, const float* wgate,
                               unsigned short* Watt, unsigned short* Wdec,
                               unsigned short* Wq, unsigned short* Wm,
                               unsigned short* Wp1, unsigned short* Wp2,
                               unsigned short* Wpr) {
  const size_t S0 = (size_t)KATT * NGATE;            // att gates
  const size_t S1 = S0 + (size_t)KDEC * NGATE;       // dec gates
  const size_t S2 = S1 + (size_t)kARNN * kATT;       // wq
  const size_t S3 = S2 + (size_t)kENC * kATT;        // wm
  const size_t S4 = S3 + (size_t)96 * kPRE;          // prenet w1 (padded K)
  const size_t S5 = S4 + (size_t)kPRE * kPRE;        // prenet w2
  const size_t S6 = S5 + (size_t)1536 * NPROJ;       // wproj|wgate|pad
  size_t i = (size_t)blockIdx.x * blockDim.x + threadIdx.x;
  size_t stride = (size_t)gridDim.x * blockDim.x;
  for (; i < S6; i += stride) {
    if (i < S0) {
      int k = (int)(i / NGATE), n = (int)(i % NGATE);
      float v = (k < 768) ? wih_a[(size_t)k * NGATE + n]
                          : whh_a[(size_t)(k - 768) * NGATE + n];
      Watt[bpack_off(k, n, KC_ATT)] = f2bf(v);
    } else if (i < S1) {
      size_t j = i - S0;
      int k = (int)(j / NGATE), n = (int)(j % NGATE);
      float v = (k < 1536) ? wih_d[(size_t)k * NGATE + n]
                           : whh_d[(size_t)(k - 1536) * NGATE + n];
      Wdec[bpack_off(k, n, KC_DEC)] = f2bf(v);
    } else if (i < S2) {
      size_t j = i - S1;
      int k = (int)(j / kATT), n = (int)(j % kATT);
      Wq[bpack_off(k, n, KC_Q)] = f2bf(wq[(size_t)k * kATT + n]);
    } else if (i < S3) {
      size_t j = i - S2;
      int k = (int)(j / kATT), n = (int)(j % kATT);
      Wm[bpack_off(k, n, KC_PM)] = f2bf(wm[(size_t)k * kATT + n]);
    } else if (i < S4) {
      size_t j = i - S3;
      int k = (int)(j / kPRE), n = (int)(j % kPRE);
      float v = (k < kNMEL) ? pw1[(size_t)k * kPRE + n] : 0.0f;
      Wp1[bpack_off(k, n, KC_P1)] = f2bf(v);
    } else if (i < S5) {
      size_t j = i - S4;
      int k = (int)(j / kPRE), n = (int)(j % kPRE);
      Wp2[bpack_off(k, n, KC_P2)] = f2bf(pw2[(size_t)k * kPRE + n]);
    } else {
      size_t j = i - S5;
      int k = (int)(j / NPROJ), n = (int)(j % NPROJ);
      float v = 0.0f;
      if (n < kNMEL) v = wproj[(size_t)k * kNMEL + n];
      else if (n == kNMEL) v = wgate[k];
      Wpr[bpack_off(k, n, KC_PROJ)] = f2bf(v);
    }
  }
}

__global__ void k_pack_memA(const float* memory, unsigned short* MemA) {
  size_t total = (size_t)MPM * kENC;
  size_t i = (size_t)blockIdx.x * blockDim.x + threadIdx.x;
  size_t stride = (size_t)gridDim.x * blockDim.x;
  for (; i < total; i += stride) {
    int m = (int)(i / kENC), k = (int)(i % kENC);
    MemA[apack_off(m, k, MT_PM)] = f2bf(memory[i]);
  }
}

__global__ void k_pack_preA(const float* dec_in, unsigned short* PreA) {
  size_t total = (size_t)MPRE * 96;
  size_t i = (size_t)blockIdx.x * blockDim.x + threadIdx.x;
  size_t stride = (size_t)gridDim.x * blockDim.x;
  for (; i < total; i += stride) {
    int m = (int)(i / 96), k = (int)(i % 96);
    int s = m / kB, b = m % kB;
    float v = 0.0f;
    if (s > 0 && k < kNMEL)
      v = dec_in[(size_t)b * kNMEL * kTOUT + (size_t)k * kTOUT + (s - 1)];
    PreA[apack_off(m, k, MT_PRE)] = f2bf(v);
  }
}

// prenet GEMM1: h1 = relu(PreA @ Wp1), output re-packed as A for GEMM2
__global__ void k_gemm_pre1(const unsigned short* A, const unsigned short* W,
                            unsigned short* H1) {
  int lane = threadIdx.x & 31;
  int gw = (int)((blockIdx.x * blockDim.x + threadIdx.x) >> 5);
  int nw = (int)((gridDim.x * blockDim.x) >> 5);
  for (int tile = gw; tile < MT_PRE * NT_P1; tile += nw) {
    int mt = tile % MT_PRE, nt = tile / MT_PRE;
    v8f acc = {};
    for (int kc = 0; kc < KC_P1; ++kc) {
      v16bf a = ldfrag(A + (((size_t)kc * MT_PRE + mt) * 32 + lane) * 16);
      v16bf b = ldfrag(W + (((size_t)nt * KC_P1 + kc) * 32 + lane) * 16);
      acc = wmma_bf16(acc, a, b);
    }
    int n = nt * 16 + (lane & 15);
    int mb = mt * 16 + (lane >> 4) * 8;
#pragma unroll
    for (int r = 0; r < 8; ++r)
      H1[apack_off(mb + r, n, MT_PRE)] = f2bf(fmaxf(acc[r], 0.0f));
  }
}

// prenet GEMM2: x = relu(H1 @ Wp2), output packed per-step into Xpack
__global__ void k_gemm_pre2(const unsigned short* A, const unsigned short* W,
                            unsigned short* Xp) {
  int lane = threadIdx.x & 31;
  int gw = (int)((blockIdx.x * blockDim.x + threadIdx.x) >> 5);
  int nw = (int)((gridDim.x * blockDim.x) >> 5);
  for (int tile = gw; tile < MT_PRE * NT_P2; tile += nw) {
    int mt = tile % MT_PRE, nt = tile / MT_PRE;
    v8f acc = {};
    for (int kc = 0; kc < KC_P2; ++kc) {
      v16bf a = ldfrag(A + (((size_t)kc * MT_PRE + mt) * 32 + lane) * 16);
      v16bf b = ldfrag(W + (((size_t)nt * KC_P2 + kc) * 32 + lane) * 16);
      acc = wmma_bf16(acc, a, b);
    }
    int n = nt * 16 + (lane & 15);
    int mb = mt * 16 + (lane >> 4) * 8;
#pragma unroll
    for (int r = 0; r < 8; ++r) {
      int m = mb + r;
      int s = m / kB, ml = m % kB;
      Xp[(size_t)s * XSTEP + apack_off(ml, n, 4)] = f2bf(fmaxf(acc[r], 0.0f));
    }
  }
}

// processed_memory = memory @ wm  (f32 out, per (b,t) row)
__global__ void k_gemm_pm(const unsigned short* A, const unsigned short* W,
                          float* pm) {
  int lane = threadIdx.x & 31;
  int gw = (int)((blockIdx.x * blockDim.x + threadIdx.x) >> 5);
  int nw = (int)((gridDim.x * blockDim.x) >> 5);
  for (int tile = gw; tile < MT_PM * NT_PM; tile += nw) {
    int mt = tile % MT_PM, nt = tile / MT_PM;
    v8f acc = {};
    for (int kc = 0; kc < KC_PM; ++kc) {
      v16bf a = ldfrag(A + (((size_t)kc * MT_PM + mt) * 32 + lane) * 16);
      v16bf b = ldfrag(W + (((size_t)nt * KC_PM + kc) * 32 + lane) * 16);
      acc = wmma_bf16(acc, a, b);
    }
    int n = nt * 16 + (lane & 15);
    int mb = mt * 16 + (lane >> 4) * 8;
#pragma unroll
    for (int r = 0; r < 8; ++r) pm[(size_t)(mb + r) * kATT + n] = acc[r];
  }
}

// ---------------- persistent decoder ----------------
struct DecArgs {
  const unsigned short* Xpack;
  const unsigned short* Watt;
  const unsigned short* Wdec;
  const unsigned short* Wq;
  const unsigned short* Wpr;
  unsigned short* Aatt;   // rel-k 0..1535 (= k-256): ctx | ah   (48 kchunks)
  unsigned short* Adec;   // k 0..2559: ah | ctx | dh            (80 kchunks)
  unsigned short* Apr;    // k 0..1535: dh | ctx                 (48 kchunks)
  float* g_att; float* g_dec;
  float* ah; float* ac; float* dh; float* dc;
  float* aw; float* awc;
  float* qproj; float* loc32; float* ebuf;
  const float* pm;
  const float* memory;
  const int* memlen;
  const float* att_b; const float* dec_b;
  const float* wconv; const float* wld; const float* vvec;
  const float* bproj; const float* bgate;
  float* out_mel; float* out_gate; float* out_align;
  unsigned* sync_cnt; unsigned* sync_gen;
};

__device__ __forceinline__ void gsync(unsigned* cnt, unsigned* gen) {
  __syncthreads();
  if (threadIdx.x == 0) {
    __threadfence();
    unsigned g = atomicAdd(gen, 0u);
    unsigned old = atomicAdd(cnt, 1u);
    if (old == (unsigned)(NBLK - 1)) {
      atomicExch(cnt, 0u);
      atomicAdd(gen, 1u);
    } else {
      while (atomicAdd(gen, 0u) == g) { __builtin_amdgcn_s_sleep(1); }
    }
    __threadfence();
  }
  __syncthreads();
}

__global__ void __launch_bounds__(NTHR, 1) k_decoder(DecArgs P) {
  __shared__ float swA[kNF * kKS];
  __shared__ float swC[kNF * kKS];
  __shared__ float swld[kNF * kATT];
  __shared__ float svv[kATT];
  __shared__ float sred[NTHR];

  for (int i = threadIdx.x; i < kNF * kKS; i += NTHR) {
    int f = i / kKS, tap = i % kKS;
    swA[i] = P.wconv[(size_t)f * 2 * kKS + tap];
    swC[i] = P.wconv[(size_t)f * 2 * kKS + kKS + tap];
  }
  for (int i = threadIdx.x; i < kNF * kATT; i += NTHR) swld[i] = P.wld[i];
  for (int i = threadIdx.x; i < kATT; i += NTHR) svv[i] = P.vvec[i];
  __syncthreads();

  const int lane = threadIdx.x & 31;
  const int gw = (int)((blockIdx.x * NTHR + threadIdx.x) >> 5);
  const int tid = (int)(blockIdx.x * NTHR + threadIdx.x);
  const int nthreads = NBLK * NTHR;

  for (int step = 0; step < kTOUT; ++step) {
    // ---- Phase 1: attention-LSTM gate GEMM: g_att = [x|ctx|ah] @ Watt ----
    {
      const unsigned short* Xs = P.Xpack + (size_t)step * XSTEP;
      for (int tile = gw; tile < 4 * NT_GATE; tile += NWAVE) {
        int mt = tile & 3, nt = tile >> 2;
        v8f acc = {};
        for (int kc = 0; kc < KC_ATT; ++kc) {
          const unsigned short* ap =
              (kc < 8) ? (Xs + (((size_t)kc * 4 + mt) * 32 + lane) * 16)
                       : (P.Aatt + (((size_t)(kc - 8) * 4 + mt) * 32 + lane) * 16);
          v16bf a = ldfrag(ap);
          v16bf b = ldfrag(P.Watt + (((size_t)nt * KC_ATT + kc) * 32 + lane) * 16);
          acc = wmma_bf16(acc, a, b);
        }
        int n = nt * 16 + (lane & 15);
        int mb = mt * 16 + (lane >> 4) * 8;
#pragma unroll
        for (int r = 0; r < 8; ++r) P.g_att[(size_t)(mb + r) * NGATE + n] = acc[r];
      }
    }
    gsync(P.sync_cnt, P.sync_gen);

    // ---- Phase 2: attention-LSTM elementwise; pack ah ----
    for (int idx = tid; idx < kB * kARNN; idx += nthreads) {
      int m = idx >> 10, j = idx & 1023;
      const float* g = P.g_att + (size_t)m * NGATE;
      float gi = g[j]        + P.att_b[j];
      float gf = g[1024 + j] + P.att_b[1024 + j];
      float gg = g[2048 + j] + P.att_b[2048 + j];
      float go = g[3072 + j] + P.att_b[3072 + j];
      float c = sigf(gf) * P.ac[idx] + sigf(gi) * tanhf(gg);
      float h = sigf(go) * tanhf(c);
      P.ac[idx] = c;
      P.ah[idx] = h;
      unsigned short hb = f2bf(h);
      P.Aatt[apack_off(m, 512 + j, 4)] = hb;  // for next step's att GEMM (k=768+j)
      P.Adec[apack_off(m, j, 4)] = hb;        // this step's dec GEMM + qproj
    }
    gsync(P.sync_cnt, P.sync_gen);

    // ---- Phase 3: qproj = ah @ wq (WMMA) + location conv (VALU) ----
    for (int tile = gw; tile < 4 * NT_Q; tile += NWAVE) {
      int mt = tile & 3, nt = tile >> 2;
      v8f acc = {};
      for (int kc = 0; kc < KC_Q; ++kc) {
        v16bf a = ldfrag(P.Adec + (((size_t)kc * 4 + mt) * 32 + lane) * 16);
        v16bf b = ldfrag(P.Wq + (((size_t)nt * KC_Q + kc) * 32 + lane) * 16);
        acc = wmma_bf16(acc, a, b);
      }
      int n = nt * 16 + (lane & 15);
      int mb = mt * 16 + (lane >> 4) * 8;
#pragma unroll
      for (int r = 0; r < 8; ++r) P.qproj[(size_t)(mb + r) * kATT + n] = acc[r];
    }
    for (int p = tid; p < kB * kTIN; p += nthreads) {
      int b = p >> 8, t = p & 255;
      float accf[kNF];
#pragma unroll
      for (int f = 0; f < kNF; ++f) accf[f] = 0.0f;
      const float* awr = P.aw + (size_t)b * kTIN;
      const float* awcr = P.awc + (size_t)b * kTIN;
      for (int tap = 0; tap < kKS; ++tap) {
        int tt = t + tap - kPAD;
        if (tt < 0 || tt >= kTIN) continue;
        float a0 = awr[tt], a1 = awcr[tt];
#pragma unroll
        for (int f = 0; f < kNF; ++f)
          accf[f] += swA[f * kKS + tap] * a0 + swC[f * kKS + tap] * a1;
      }
      float* lo = P.loc32 + (size_t)p * kNF;
#pragma unroll
      for (int f = 0; f < kNF; ++f) lo[f] = accf[f];
    }
    gsync(P.sync_cnt, P.sync_gen);

    // ---- Phase 4: energies e[b,t] = v . tanh(q + loc@wld + pm), masked ----
    for (int p = tid; p < kB * kTIN; p += nthreads) {
      int b = p >> 8, t = p & 255;
      const float* lo = P.loc32 + (size_t)p * kNF;
      const float* pmr = P.pm + (size_t)p * kATT;
      const float* qr = P.qproj + (size_t)b * kATT;
      float lof[kNF];
#pragma unroll
      for (int f = 0; f < kNF; ++f) lof[f] = lo[f];
      float s = 0.0f;
      for (int i = 0; i < kATT; ++i) {
        float l = qr[i] + pmr[i];
#pragma unroll
        for (int f = 0; f < kNF; ++f) l += lof[f] * swld[f * kATT + i];
        s += svv[i] * tanhf(l);
      }
      P.ebuf[p] = (t >= P.memlen[b]) ? -1.0e9f : s;
    }
    gsync(P.sync_cnt, P.sync_gen);

    // ---- Phase 5: softmax + context; pack ctx; emit alignments ----
    for (int bb = 0; bb < 2; ++bb) {
      int b = (int)blockIdx.x * 2 + bb;
      int t = threadIdx.x;  // NTHR == kTIN
      float ev = P.ebuf[(size_t)b * kTIN + t];
      sred[t] = ev; __syncthreads();
      for (int off = 128; off > 0; off >>= 1) {
        if (t < off) sred[t] = fmaxf(sred[t], sred[t + off]);
        __syncthreads();
      }
      float mx = sred[0]; __syncthreads();
      float ex = __expf(ev - mx);
      sred[t] = ex; __syncthreads();
      for (int off = 128; off > 0; off >>= 1) {
        if (t < off) sred[t] += sred[t + off];
        __syncthreads();
      }
      float sum = sred[0]; __syncthreads();
      float w = ex / sum;
      P.aw[(size_t)b * kTIN + t] = w;
      P.awc[(size_t)b * kTIN + t] += w;
      P.out_align[((size_t)b * kTOUT + step) * kTIN + t] = w;
      sred[t] = w; __syncthreads();
      for (int d = t; d < kENC; d += NTHR) {
        float s = 0.0f;
        const float* mem = P.memory + (size_t)b * kTIN * kENC + d;
        for (int tt = 0; tt < kTIN; ++tt) s += sred[tt] * mem[(size_t)tt * kENC];
        unsigned short cb = f2bf(s);
        P.Aatt[apack_off(b, d, 4)] = cb;          // next step att (k=256+d)
        P.Adec[apack_off(b, 1024 + d, 4)] = cb;   // this step dec
        P.Apr[apack_off(b, 1024 + d, 4)] = cb;    // this step proj
      }
      __syncthreads();
    }
    gsync(P.sync_cnt, P.sync_gen);

    // ---- Phase 6: decoder-LSTM gate GEMM: g_dec = [ah|ctx|dh] @ Wdec ----
    for (int tile = gw; tile < 4 * NT_GATE; tile += NWAVE) {
      int mt = tile & 3, nt = tile >> 2;
      v8f acc = {};
      for (int kc = 0; kc < KC_DEC; ++kc) {
        v16bf a = ldfrag(P.Adec + (((size_t)kc * 4 + mt) * 32 + lane) * 16);
        v16bf b = ldfrag(P.Wdec + (((size_t)nt * KC_DEC + kc) * 32 + lane) * 16);
        acc = wmma_bf16(acc, a, b);
      }
      int n = nt * 16 + (lane & 15);
      int mb = mt * 16 + (lane >> 4) * 8;
#pragma unroll
      for (int r = 0; r < 8; ++r) P.g_dec[(size_t)(mb + r) * NGATE + n] = acc[r];
    }
    gsync(P.sync_cnt, P.sync_gen);

    // ---- Phase 7: decoder-LSTM elementwise; pack dh ----
    for (int idx = tid; idx < kB * kARNN; idx += nthreads) {
      int m = idx >> 10, j = idx & 1023;
      const float* g = P.g_dec + (size_t)m * NGATE;
      float gi = g[j]        + P.dec_b[j];
      float gf = g[1024 + j] + P.dec_b[1024 + j];
      float gg = g[2048 + j] + P.dec_b[2048 + j];
      float go = g[3072 + j] + P.dec_b[3072 + j];
      float c = sigf(gf) * P.dc[idx] + sigf(gi) * tanhf(gg);
      float h = sigf(go) * tanhf(c);
      P.dc[idx] = c;
      P.dh[idx] = h;
      unsigned short hb = f2bf(h);
      P.Adec[apack_off(m, 1536 + j, 4)] = hb;  // next step dec GEMM
      P.Apr[apack_off(m, j, 4)] = hb;          // this step proj
    }
    gsync(P.sync_cnt, P.sync_gen);

    // ---- Phase 8: projection GEMM: [dh|ctx] @ [wproj|wgate] ----
    for (int tile = gw; tile < 4 * NT_PROJ; tile += NWAVE) {
      int mt = tile & 3, nt = tile >> 2;
      v8f acc = {};
      for (int kc = 0; kc < KC_PROJ; ++kc) {
        v16bf a = ldfrag(P.Apr + (((size_t)kc * 4 + mt) * 32 + lane) * 16);
        v16bf b = ldfrag(P.Wpr + (((size_t)nt * KC_PROJ + kc) * 32 + lane) * 16);
        acc = wmma_bf16(acc, a, b);
      }
      int n = nt * 16 + (lane & 15);
      int mb = mt * 16 + (lane >> 4) * 8;
#pragma unroll
      for (int r = 0; r < 8; ++r) {
        int m = mb + r;
        if (n < kNMEL) {
          P.out_mel[(size_t)m * kNMEL * kTOUT + (size_t)n * kTOUT + step] =
              acc[r] + P.bproj[n];
        } else if (n == kNMEL) {
          P.out_gate[(size_t)m * kTOUT + step] = acc[r] + P.bgate[0];
        }
      }
    }
    gsync(P.sync_cnt, P.sync_gen);
  }
}

// ---------------- host launch ----------------
extern "C" void kernel_launch(void* const* d_in, const int* in_sizes, int n_in,
                              void* d_out, int out_size, void* d_ws, size_t ws_size,
                              hipStream_t stream) {
  (void)in_sizes; (void)n_in; (void)out_size; (void)ws_size;

  const float* memory   = (const float*)d_in[0];
  const float* dec_in   = (const float*)d_in[1];
  const int*   memlen   = (const int*)d_in[2];
  const float* pw1      = (const float*)d_in[3];
  const float* pw2      = (const float*)d_in[4];
  const float* att_wih  = (const float*)d_in[5];
  const float* att_whh  = (const float*)d_in[6];
  const float* att_b    = (const float*)d_in[7];
  const float* wq       = (const float*)d_in[8];
  const float* wm       = (const float*)d_in[9];
  const float* vvec     = (const float*)d_in[10];
  const float* wconv    = (const float*)d_in[11];
  const float* wld      = (const float*)d_in[12];
  const float* dec_wih  = (const float*)d_in[13];
  const float* dec_whh  = (const float*)d_in[14];
  const float* dec_b    = (const float*)d_in[15];
  const float* wproj    = (const float*)d_in[16];
  const float* bproj    = (const float*)d_in[17];
  const float* wgate    = (const float*)d_in[18];
  const float* bgate    = (const float*)d_in[19];

  char* base = (char*)d_ws;
  size_t off = 0;
  auto alloc = [&](size_t bytes) -> void* {
    off = (off + 255) & ~(size_t)255;
    void* p = base + off;
    off += bytes;
    return p;
  };

  // ---- zero-initialized region (state carried across steps) ----
  unsigned* sync_cnt = (unsigned*)alloc(8);
  unsigned* sync_gen = sync_cnt + 1;
  float* aw  = (float*)alloc((size_t)kB * kTIN * 4);
  float* awc = (float*)alloc((size_t)kB * kTIN * 4);
  float* ah  = (float*)alloc((size_t)kB * kARNN * 4);
  float* ac  = (float*)alloc((size_t)kB * kARNN * 4);
  float* dh  = (float*)alloc((size_t)kB * kARNN * 4);
  float* dc  = (float*)alloc((size_t)kB * kARNN * 4);
  unsigned short* Aatt = (unsigned short*)alloc((size_t)48 * 2048 * 2);
  unsigned short* Adec = (unsigned short*)alloc((size_t)80 * 2048 * 2);
  unsigned short* Apr  = (unsigned short*)alloc((size_t)48 * 2048 * 2);
  size_t zero_bytes = (off + 255) & ~(size_t)255;

  // ---- scratch / packed constants ----
  float* g_att  = (float*)alloc((size_t)kB * NGATE * 4);
  float* g_dec  = (float*)alloc((size_t)kB * NGATE * 4);
  float* qproj  = (float*)alloc((size_t)kB * kATT * 4);
  float* loc32  = (float*)alloc((size_t)kB * kTIN * kNF * 4);
  float* ebuf   = (float*)alloc((size_t)kB * kTIN * 4);
  float* pm     = (float*)alloc((size_t)MPM * kATT * 4);
  unsigned short* Watt = (unsigned short*)alloc((size_t)NT_GATE * KC_ATT * 512 * 2);
  unsigned short* Wdec = (unsigned short*)alloc((size_t)NT_GATE * KC_DEC * 512 * 2);
  unsigned short* Wq   = (unsigned short*)alloc((size_t)NT_Q * KC_Q * 512 * 2);
  unsigned short* Wm   = (unsigned short*)alloc((size_t)NT_PM * KC_PM * 512 * 2);
  unsigned short* Wp1  = (unsigned short*)alloc((size_t)NT_P1 * KC_P1 * 512 * 2);
  unsigned short* Wp2  = (unsigned short*)alloc((size_t)NT_P2 * KC_P2 * 512 * 2);
  unsigned short* Wpr  = (unsigned short*)alloc((size_t)NT_PROJ * KC_PROJ * 512 * 2);
  unsigned short* Xpack = (unsigned short*)alloc((size_t)kTOUT * XSTEP * 2);
  unsigned short* MemA  = (unsigned short*)alloc((size_t)KC_PM * MT_PM * 512 * 2);
  unsigned short* PreA  = (unsigned short*)alloc((size_t)KC_P1 * MT_PRE * 512 * 2);
  unsigned short* H1    = (unsigned short*)alloc((size_t)KC_P2 * MT_PRE * 512 * 2);

  float* out_mel   = (float*)d_out;
  float* out_gate  = out_mel + (size_t)kB * kNMEL * kTOUT;
  float* out_align = out_gate + (size_t)kB * kTOUT;

  // ---- setup ----
  k_zero<<<512, 256, 0, stream>>>((float*)d_ws, zero_bytes / 4);
  k_pack_weights<<<2048, 256, 0, stream>>>(att_wih, att_whh, dec_wih, dec_whh,
                                           wq, wm, pw1, pw2, wproj, wgate,
                                           Watt, Wdec, Wq, Wm, Wp1, Wp2, Wpr);
  k_pack_memA<<<2048, 256, 0, stream>>>(memory, MemA);
  k_pack_preA<<<2048, 256, 0, stream>>>(dec_in, PreA);
  k_gemm_pre1<<<2048, 256, 0, stream>>>(PreA, Wp1, H1);
  k_gemm_pre2<<<2048, 256, 0, stream>>>(H1, Wp2, Xpack);
  k_gemm_pm<<<1024, 256, 0, stream>>>(MemA, Wm, pm);

  // ---- persistent sequential decoder ----
  DecArgs P;
  P.Xpack = Xpack; P.Watt = Watt; P.Wdec = Wdec; P.Wq = Wq; P.Wpr = Wpr;
  P.Aatt = Aatt; P.Adec = Adec; P.Apr = Apr;
  P.g_att = g_att; P.g_dec = g_dec;
  P.ah = ah; P.ac = ac; P.dh = dh; P.dc = dc;
  P.aw = aw; P.awc = awc;
  P.qproj = qproj; P.loc32 = loc32; P.ebuf = ebuf;
  P.pm = pm; P.memory = memory; P.memlen = memlen;
  P.att_b = att_b; P.dec_b = dec_b;
  P.wconv = wconv; P.wld = wld; P.vvec = vvec;
  P.bproj = bproj; P.bgate = bgate;
  P.out_mel = out_mel; P.out_gate = out_gate; P.out_align = out_align;
  P.sync_cnt = sync_cnt; P.sync_gen = sync_gen;
  k_decoder<<<NBLK, NTHR, 0, stream>>>(P);
}
